// ResBlock_3d_25125558681985
// MI455X (gfx1250) — matrix operand order repaired and checked by reading, not compiled, not run on hardware
//
#include <hip/hip_runtime.h>
#include <hip/hip_bf16.h>
#include <stdint.h>

// ---------------------------------------------------------------------------
// DCN-3D residual block for MI455X (gfx1250, wave32).
// Both convs (offset conv + main deformable conv) run as implicit GEMMs on
// v_wmma_f32_16x16x32_bf16 with fp32 accumulation. Deformable bilinear
// sampling is done in VALU into an LDS A-tile per 3x3x3 tap. Weight (B)
// tap-blocks are double-buffered into LDS via gfx1250 async load-to-LDS.
// ---------------------------------------------------------------------------

typedef __attribute__((ext_vector_type(16))) __bf16 v16bf;
typedef __attribute__((ext_vector_type(8)))  float  v8f;
typedef __attribute__((ext_vector_type(4)))  int    v4i;

#define Tdim 7
#define Hdim 48
#define Wdim 48
#define Cdim 64
#define Mpix (Tdim*Hdim*Wdim)   // 16128 output pixels
#define KK   27                 // 3x3x3 taps
#define LDA  66                 // padded LDS row stride in shorts (bank spread)

#if __has_builtin(__builtin_amdgcn_global_load_async_to_lds_b128)
#define HAVE_ASYNC_LDS 1
#endif

__device__ __forceinline__ unsigned short f2bf(float f){
  unsigned int u = __float_as_uint(f);
  u += 0x7FFFu + ((u >> 16) & 1u);     // round-to-nearest-even
  return (unsigned short)(u >> 16);
}
__device__ __forceinline__ float bf2f(unsigned short h){
  return __uint_as_float(((unsigned int)h) << 16);
}

// 16-byte global -> LDS copy: async DMA when available, else through VGPRs.
__device__ __forceinline__ void copy16_g2l(const void* gsrc, void* ldst){
#ifdef HAVE_ASYNC_LDS
  __builtin_amdgcn_global_load_async_to_lds_b128(
      (v4i*)(uintptr_t)gsrc, (v4i*)(uintptr_t)ldst, 0, 0);
#else
  *(uint4*)ldst = *(const uint4*)gsrc;
#endif
}

template <int N>
__device__ __forceinline__ void wait_async(){
#ifdef HAVE_ASYNC_LDS
#if __has_builtin(__builtin_amdgcn_s_wait_asynccnt)
  __builtin_amdgcn_s_wait_asynccnt(N);
#else
  asm volatile("s_wait_asynccnt %0" :: "i"(N) : "memory");
#endif
#endif
}

// x (C, T*H*W) fp32  ->  xbf (T*H*W, C) bf16 channels-last
__global__ void k_cvt_x(const float* __restrict__ x, unsigned short* __restrict__ xbf){
  int idx = blockIdx.x * 256 + threadIdx.x;        // [0, Mpix*64)
  int p = idx >> 6, c = idx & 63;
  xbf[idx] = f2bf(x[(size_t)c * Mpix + p]);
}

// Swizzle weights (O, 64, 27) fp32 into WMMA-B fragment order:
// flat = tap*4096 + ks*2048 + ntile*512 + lane*16 + e, bf16, O padded to 64.
__global__ void k_cvt_w(const float* __restrict__ w, int O, unsigned short* __restrict__ dst){
  int idx  = blockIdx.x * 256 + threadIdx.x;       // [0, 27*2*4*32*16 = 110592)
  int e    = idx & 15;
  int lane = (idx >> 4) & 31;
  int nt   = (idx >> 9) & 3;
  int ks   = (idx >> 11) & 1;
  int tap  = idx >> 12;
  int klocal = (e < 8) ? ((lane >> 4) * 8 + e) : (16 + (lane >> 4) * 8 + (e - 8));  // B 32x16 bf16 layout
  int c = ks * 32 + klocal;
  int o = nt * 16 + (lane & 15);
  float v = (o < O) ? w[((size_t)o * Cdim + c) * KK + tap] : 0.f;
  dst[idx] = f2bf(v);
}

// One workgroup: 64-pixel M-tile x 64 output channels. 128 threads = 4 wave32.
// Wave w computes rows [16w,16w+16) x all 64 N via 4 accumulator tiles.
// mode 0: write f32 offsets (M,64).  mode 1: +bias, LeakyReLU -> bf16 (M,64).
// mode 2: +bias +residual -> f32 (64,M) (= NCTHW for N=1).
__global__ void __launch_bounds__(128)
k_dcn_gemm(const unsigned short* __restrict__ xin,   // (Mpix,64) bf16
           const unsigned short* __restrict__ wmat,  // swizzled B fragments bf16
           const float* __restrict__ bias,           // (NO) f32
           const float* __restrict__ offb,           // (Mpix,64) f32, deform only
           const float* __restrict__ resid,          // (64,Mpix) f32, mode 2
           float* __restrict__ outF,
           unsigned short* __restrict__ outB,
           int NO, int mode, int deform)
{
  __shared__ unsigned short Atile[64 * LDA];
  __shared__ unsigned short Btile[2][64 * 64];      // 8 KB per tap-block

  const int tid = threadIdx.x;
  const int m0  = blockIdx.x * 64;
  const int wv  = tid >> 5;
  const int ln  = tid & 31;

  // sampling assignment: 2 threads per pixel, 32 channels each
  const int p_local = tid >> 1;
  const int cbase   = (tid & 1) * 32;
  const int p  = m0 + p_local;
  const int w_ = p % Wdim;
  const int h_ = (p / Wdim) % Hdim;
  const int t_ = p / (Wdim * Hdim);

  v8f acc[4];
  const v8f vzero = {0.f,0.f,0.f,0.f,0.f,0.f,0.f,0.f};
  #pragma unroll
  for (int j = 0; j < 4; ++j) acc[j] = vzero;

  // prologue: stage tap 0's 8KB weight block into Btile[0]
  {
    const char* src = (const char*)wmat + tid * 64;
    char* dst = (char*)&Btile[0][0] + tid * 64;
    #pragma unroll
    for (int i = 0; i < 4; ++i) copy16_g2l(src + i * 16, dst + i * 16);
  }

  int kt = 0, kh = 0, kw = 0;                      // carried tap decode
  for (int tap = 0; tap < KK; ++tap){
    const int bp = tap & 1;
    const int pt  = t_ + kt - 1;
    const bool tok = (pt >= 0) && (pt < Tdim);
    const int ptc = pt < 0 ? 0 : (pt > Tdim - 1 ? Tdim - 1 : pt);

    // 4 corner pointers + bilinear weights (clamped addr, zeroed weight => no
    // divergence: EXEC stays all-ones into the WMMA section).
    const unsigned short* cp[4];
    float cw[4];
    if (deform){
      const float oh = offb[(size_t)p * 64 + tap * 2 + 0];
      const float ow = offb[(size_t)p * 64 + tap * 2 + 1];
      const float ph = (float)(h_ + kh - 1) + oh;
      const float pw = (float)(w_ + kw - 1) + ow;
      const float h0f = floorf(ph), w0f = floorf(pw);
      const float lh = ph - h0f, lw = pw - w0f;
      const int h0 = (int)h0f, w0 = (int)w0f;
      const float wq[4] = {(1.f - lh) * (1.f - lw), (1.f - lh) * lw,
                           lh * (1.f - lw),         lh * lw};
      #pragma unroll
      for (int cn = 0; cn < 4; ++cn){
        const int hc = h0 + (cn >> 1);
        const int wc = w0 + (cn & 1);
        const bool ok = tok && hc >= 0 && hc < Hdim && wc >= 0 && wc < Wdim;
        const int hcc = hc < 0 ? 0 : (hc > Hdim - 1 ? Hdim - 1 : hc);
        const int wcc = wc < 0 ? 0 : (wc > Wdim - 1 ? Wdim - 1 : wc);
        cp[cn] = xin + (((size_t)ptc * Hdim + hcc) * Wdim + wcc) * 64 + cbase;
        cw[cn] = ok ? wq[cn] : 0.f;
      }
    } else {
      const int hc = h_ + kh - 1, wc = w_ + kw - 1;
      const bool ok = tok && hc >= 0 && hc < Hdim && wc >= 0 && wc < Wdim;
      const int hcc = hc < 0 ? 0 : (hc > Hdim - 1 ? Hdim - 1 : hc);
      const int wcc = wc < 0 ? 0 : (wc > Wdim - 1 ? Wdim - 1 : wc);
      cp[0] = xin + (((size_t)ptc * Hdim + hcc) * Wdim + wcc) * 64 + cbase;
      cw[0] = ok ? 1.f : 0.f;
    }

    // prefetch next tap's weight block into the other LDS buffer (async DMA)
    if (tap + 1 < KK){
      const char* src = (const char*)wmat + (size_t)(tap + 1) * 8192 + tid * 64;
      char* dst = (char*)&Btile[bp ^ 1][0] + tid * 64;
      #pragma unroll
      for (int i = 0; i < 4; ++i) copy16_g2l(src + i * 16, dst + i * 16);
    }

    // build A tile: 8 channels per group (uniform branch on deform)
    unsigned int* drow = (unsigned int*)&Atile[p_local * LDA + cbase];
    #pragma unroll
    for (int g = 0; g < 4; ++g){
      float a8[8];
      #pragma unroll
      for (int i = 0; i < 8; ++i) a8[i] = 0.f;
      if (deform){
        #pragma unroll
        for (int cn = 0; cn < 4; ++cn){
          const uint4 q = ((const uint4*)cp[cn])[g];
          const float wvq = cw[cn];
          a8[0] += wvq * bf2f((unsigned short)(q.x & 0xFFFFu));
          a8[1] += wvq * bf2f((unsigned short)(q.x >> 16));
          a8[2] += wvq * bf2f((unsigned short)(q.y & 0xFFFFu));
          a8[3] += wvq * bf2f((unsigned short)(q.y >> 16));
          a8[4] += wvq * bf2f((unsigned short)(q.z & 0xFFFFu));
          a8[5] += wvq * bf2f((unsigned short)(q.z >> 16));
          a8[6] += wvq * bf2f((unsigned short)(q.w & 0xFFFFu));
          a8[7] += wvq * bf2f((unsigned short)(q.w >> 16));
        }
      } else {
        const uint4 q = ((const uint4*)cp[0])[g];
        const float wvq = cw[0];
        a8[0] = wvq * bf2f((unsigned short)(q.x & 0xFFFFu));
        a8[1] = wvq * bf2f((unsigned short)(q.x >> 16));
        a8[2] = wvq * bf2f((unsigned short)(q.y & 0xFFFFu));
        a8[3] = wvq * bf2f((unsigned short)(q.y >> 16));
        a8[4] = wvq * bf2f((unsigned short)(q.z & 0xFFFFu));
        a8[5] = wvq * bf2f((unsigned short)(q.z >> 16));
        a8[6] = wvq * bf2f((unsigned short)(q.w & 0xFFFFu));
        a8[7] = wvq * bf2f((unsigned short)(q.w >> 16));
      }
      #pragma unroll
      for (int j2 = 0; j2 < 4; ++j2)
        drow[g * 4 + j2] = (unsigned int)f2bf(a8[2 * j2]) |
                           ((unsigned int)f2bf(a8[2 * j2 + 1]) << 16);
    }

    // current tap's B block must have landed (async loads complete in order:
    // after issuing 4 new ops, cnt<=4 implies the previous 4 are done)
    if (tap + 1 < KK) wait_async<4>(); else wait_async<0>();
    __syncthreads();

    // WMMA: this tap contributes K=64 => 2 K-steps of 32, 4 N-tiles each
    const int mrow = 16 * wv + (ln & 15);
    const int kb   = (ln >> 4) * 8;
    const unsigned short* bt = &Btile[bp][0];
    #pragma unroll
    for (int ks = 0; ks < 2; ++ks){
      union { v16bf v; unsigned int u[8]; } af;
      #pragma unroll
      for (int r = 0; r < 8; ++r){
        const int kk2 = ks * 32 + ((r < 4) ? (kb + 2 * r) : (16 + kb + 2 * (r - 4)));
        af.u[r] = *(const unsigned int*)&Atile[mrow * LDA + kk2];
      }
      #pragma unroll
      for (int j = 0; j < 4; ++j){
        union { v16bf v; uint4 q[2]; } bfv;
        const uint4* bp2 = (const uint4*)(bt + ((ks * 4 + j) * 32 + ln) * 16);
        bfv.q[0] = bp2[0];
        bfv.q[1] = bp2[1];
        acc[j] = __builtin_amdgcn_wmma_f32_16x16x32_bf16(
            false, af.v, false, bfv.v, (short)0, acc[j], false, false);
      }
    }
    __syncthreads();

    if (++kw == 3){ kw = 0; if (++kh == 3){ kh = 0; ++kt; } }
  }

  // epilogue: C/D 16x16 f32 layout -> (row = r + 8*(ln>>4), col = ln&15)
  #pragma unroll
  for (int j = 0; j < 4; ++j){
    const int o  = 16 * j + (ln & 15);
    const float bo = (o < NO) ? bias[o] : 0.f;
    #pragma unroll
    for (int r = 0; r < 8; ++r){
      float y = acc[j][r];
      const int ml = 16 * wv + r + 8 * (ln >> 4);
      const int pp = m0 + ml;
      if (mode == 0){
        outF[(size_t)pp * 64 + o] = y + bo;
      } else if (mode == 1){
        float z = y + bo;
        z = (z >= 0.f) ? z : 0.1f * z;
        outB[(size_t)pp * 64 + o] = f2bf(z);
      } else {
        outF[(size_t)o * Mpix + pp] = y + bo + resid[(size_t)o * Mpix + pp];
      }
    }
  }
}

extern "C" void kernel_launch(void* const* d_in, const int* in_sizes, int n_in,
                              void* d_out, int out_size, void* d_ws, size_t ws_size,
                              hipStream_t stream) {
  const float* x     = (const float*)d_in[0];
  const float* woff0 = (const float*)d_in[1];
  const float* boff0 = (const float*)d_in[2];
  const float* w0    = (const float*)d_in[3];
  const float* b0    = (const float*)d_in[4];
  const float* woff1 = (const float*)d_in[5];
  const float* boff1 = (const float*)d_in[6];
  const float* w1    = (const float*)d_in[7];
  const float* b1    = (const float*)d_in[8];
  float* out = (float*)d_out;

  // workspace layout (~9.2 MB)
  char* ws = (char*)d_ws;
  unsigned short* xbf  = (unsigned short*)(ws);                       // 2*M*64
  unsigned short* hbf  = (unsigned short*)(ws + 2064384);             // 2*M*64
  float*          offb = (float*)(ws + 2 * 2064384);                  // 4*M*64
  unsigned short* wb0o = (unsigned short*)(ws + 2 * 2064384 + 4128768);
  unsigned short* wb0  = wb0o + 110592;
  unsigned short* wb1o = wb0  + 110592;
  unsigned short* wb1  = wb1o + 110592;

  k_cvt_x<<<dim3(4032), dim3(256), 0, stream>>>(x, xbf);
  k_cvt_w<<<dim3(432), dim3(256), 0, stream>>>(woff0, 54, wb0o);
  k_cvt_w<<<dim3(432), dim3(256), 0, stream>>>(w0,    64, wb0);
  k_cvt_w<<<dim3(432), dim3(256), 0, stream>>>(woff1, 54, wb1o);
  k_cvt_w<<<dim3(432), dim3(256), 0, stream>>>(w1,    64, wb1);

  // layer 0: offset conv -> deformable conv (+bias, LeakyReLU) -> hbf
  k_dcn_gemm<<<dim3(252), dim3(128), 0, stream>>>(
      xbf, wb0o, boff0, nullptr, nullptr, offb, nullptr, 54, 0, 0);
  k_dcn_gemm<<<dim3(252), dim3(128), 0, stream>>>(
      xbf, wb0, b0, offb, nullptr, nullptr, hbf, 64, 1, 1);

  // layer 1: offset conv -> deformable conv (+bias, +x residual) -> out
  k_dcn_gemm<<<dim3(252), dim3(128), 0, stream>>>(
      hbf, wb1o, boff1, nullptr, nullptr, offb, nullptr, 54, 0, 0);
  k_dcn_gemm<<<dim3(252), dim3(128), 0, stream>>>(
      hbf, wb1, b1, offb, x, out, nullptr, 64, 2, 1);
}
